// CausalSelfAttention_23356032156057
// MI455X (gfx1250) — matrix operand order repaired and checked by reading, not compiled, not run on hardware
//
#include <hip/hip_runtime.h>
#include <hip/hip_bf16.h>

// ---------------------------------------------------------------------------
// Causal self-attention, MI455X (gfx1250), wave32.
//   B=4, T=2048, C=1024, H=16, D=64.
// All GEMMs (QKV, QK^T, PV, proj) on v_wmma_f32_16x16x32_bf16 with fp32 accum.
// GEMM B-tiles staged to LDS with GLOBAL_LOAD_ASYNC_TO_LDS_B128 (ASYNCcnt),
// double-buffered; bank-conflict-free padded LDS layouts.
// Workspace: 92,274,688 bytes.
// ---------------------------------------------------------------------------

#define BATCH 4
#define SEQ   2048
#define EMB   1024
#define NH    16
#define HD    64
#define MTOT  (BATCH * SEQ)   /* 8192 */
#define NQKV  (3 * EMB)       /* 3072 */
#define BLDS  72              /* padded row stride (elems) of LDS B tile */

typedef __attribute__((ext_vector_type(16))) __bf16 v16bf;
typedef __attribute__((ext_vector_type(8)))  __bf16 v8bf;
typedef __attribute__((ext_vector_type(8)))  float  v8f;
typedef int v4i_async __attribute__((vector_size(16)));   // matches builtin proto

#if defined(__has_builtin)
#if __has_builtin(__builtin_amdgcn_global_load_async_to_lds_b128)
#define HAVE_ASYNC_BUILTIN 1
#endif
#if __has_builtin(__builtin_amdgcn_s_wait_asynccnt)
#define HAVE_WAIT_BUILTIN 1
#endif
#endif

static __device__ __forceinline__ void async_copy_b128(const __bf16* g, __bf16* l) {
#ifdef HAVE_ASYNC_BUILTIN
  __builtin_amdgcn_global_load_async_to_lds_b128(
      (__attribute__((address_space(1))) v4i_async*)g,
      (__attribute__((address_space(3))) v4i_async*)l, 0, 0);
#else
  unsigned loff = (unsigned)(unsigned long long)
      (__attribute__((address_space(3))) void*)l;
  asm volatile("global_load_async_to_lds_b128 %0, %1, off"
               :: "v"(loff), "v"((unsigned long long)g) : "memory");
#endif
}

static __device__ __forceinline__ void wait_async_0() {
#ifdef HAVE_WAIT_BUILTIN
  __builtin_amdgcn_s_wait_asynccnt(0);
#else
  asm volatile("s_wait_asynccnt 0x0" ::: "memory");
#endif
}
static __device__ __forceinline__ void wait_async_4() {
#ifdef HAVE_WAIT_BUILTIN
  __builtin_amdgcn_s_wait_asynccnt(4);
#else
  asm volatile("s_wait_asynccnt 0x4" ::: "memory");
#endif
}

static __device__ __forceinline__ v8f wmma_bf16(v16bf a, v16bf b, v8f c) {
  return __builtin_amdgcn_wmma_f32_16x16x32_bf16(
      false, a, false, b, (short)0, c, false, false);
}

// A fragment: 16(M) x 32(K), row-major, row stride ld (ISA 7.12.2 16-bit A).
static __device__ __forceinline__ v16bf load_a_frag(const __bf16* base, int ld, int lane) {
  int row  = lane & 15;
  int half = lane >> 4;
  const __bf16* p = base + (size_t)row * ld + half * 8;
  v8bf lo = *reinterpret_cast<const v8bf*>(p);       // K = half*8 .. +7
  v8bf hi = *reinterpret_cast<const v8bf*>(p + 16);  // K = 16+half*8 .. +7
  v16bf r;
#pragma unroll
  for (int i = 0; i < 8; ++i) { r[i] = lo[i]; r[i + 8] = hi[i]; }
  return r;
}

// B fragment: 32(K) x 16(N); B stored N-major ([N x K], row stride ld).
static __device__ __forceinline__ v16bf load_b_frag(const __bf16* base, int ld, int lane) {
  int n    = lane & 15;
  int half = lane >> 4;
  const __bf16* p = base + (size_t)n * ld + half * 8;    // K = half*8.., 16+half*8..
  v8bf lo = *reinterpret_cast<const v8bf*>(p);
  v8bf hi = *reinterpret_cast<const v8bf*>(p + 16);
  v16bf r;
#pragma unroll
  for (int i = 0; i < 8; ++i) { r[i] = lo[i]; r[i + 8] = hi[i]; }
  return r;
}

// Row reductions inside the 16-lane half-group that owns one C/D row.
static __device__ __forceinline__ float rowred_max(float v) {
  v = fmaxf(v, __shfl_xor(v, 1, 32));
  v = fmaxf(v, __shfl_xor(v, 2, 32));
  v = fmaxf(v, __shfl_xor(v, 4, 32));
  v = fmaxf(v, __shfl_xor(v, 8, 32));
  return v;
}
static __device__ __forceinline__ float rowred_sum(float v) {
  v += __shfl_xor(v, 1, 32);
  v += __shfl_xor(v, 2, 32);
  v += __shfl_xor(v, 4, 32);
  v += __shfl_xor(v, 8, 32);
  return v;
}

// --- fp32 -> bf16 conversions (weights transposed to [N x K]) ---------------
__global__ void cvt_x_kernel(const float* __restrict__ x, __bf16* __restrict__ xb, int n) {
  int i = blockIdx.x * 256 + threadIdx.x;
  if (i < n) xb[i] = (__bf16)x[i];
}
__global__ void cvt_wqkv_kernel(const float* __restrict__ w, __bf16* __restrict__ wt) {
  int i = blockIdx.x * 256 + threadIdx.x;
  if (i < NQKV * EMB) {
    int n = i >> 10, c = i & (EMB - 1);
    wt[i] = (__bf16)w[(size_t)c * NQKV + n];
  }
}
__global__ void cvt_wproj_kernel(const float* __restrict__ w, __bf16* __restrict__ wt) {
  int i = blockIdx.x * 256 + threadIdx.x;
  if (i < EMB * EMB) {
    int n = i >> 10, c = i & (EMB - 1);
    wt[i] = (__bf16)w[(size_t)c * EMB + n];
  }
}

// ---------------------------------------------------------------------------
// GEMM core: block = 256 threads (8 waves), block tile 128(M) x 128(N),
// wave tile 32(M) x 64(N). B tile [128 x 64K] double-buffered in LDS via
// async copies (each thread moves 4 x 16B per stage, ASYNCcnt-tracked).
// ---------------------------------------------------------------------------
#define GEMM_PROLOGUE(AptrExpr, WT)                                            \
  __shared__ __attribute__((aligned(16))) __bf16 Bsh[2][128 * BLDS];           \
  int tid  = threadIdx.x;                                                      \
  int lane = tid & 31;                                                         \
  int wid  = tid >> 5;                                                         \
  int half = lane >> 4;                                                        \
  int mw = wid & 3, nw = wid >> 2;                                             \
  int mBase    = blockIdx.y * 128 + mw * 32;                                   \
  int nBaseBlk = blockIdx.x * 128;                                             \
  int nBase    = nw * 64; /* within block */                                   \
  v8f acc[2][4] = {};                                                          \
  /* stage issue: chunk c = tid + 256*j ; row = c>>3 (0..127), ch = c&7 */     \
  auto issue_stage = [&](int buf, int k0) {                                    \
    _Pragma("unroll")                                                          \
    for (int j = 0; j < 4; ++j) {                                              \
      int c = tid + 256 * j;                                                   \
      int row = c >> 3, ch = c & 7;                                            \
      const __bf16* g = WT + (size_t)(nBaseBlk + row) * EMB + k0 + ch * 8;     \
      __bf16* l = &Bsh[buf][row * BLDS + ch * 8];                              \
      async_copy_b128(g, l);                                                   \
    }                                                                          \
  };                                                                           \
  issue_stage(0, 0);                                                           \
  for (int ks = 0; ks < EMB / 64; ++ks) {                                      \
    if (ks + 1 < EMB / 64) { issue_stage((ks + 1) & 1, (ks + 1) * 64);         \
                             wait_async_4(); }                                 \
    else                   { wait_async_0(); }                                 \
    __syncthreads();                                                           \
    const __bf16* Bl = &Bsh[ks & 1][0];                                        \
    _Pragma("unroll")                                                          \
    for (int kk = 0; kk < 64; kk += 32) {                                      \
      v16bf a0 = load_a_frag((AptrExpr) + (size_t)mBase * EMB + ks * 64 + kk,  \
                             EMB, lane);                                       \
      v16bf a1 = load_a_frag((AptrExpr) + (size_t)(mBase + 16) * EMB +         \
                             ks * 64 + kk, EMB, lane);                         \
      _Pragma("unroll")                                                        \
      for (int t = 0; t < 4; ++t) {                                            \
        v16bf b = load_b_frag(Bl + (size_t)(nBase + t * 16) * BLDS + kk,       \
                              BLDS, lane);                                     \
        acc[0][t] = wmma_bf16(a0, b, acc[0][t]);                               \
        acc[1][t] = wmma_bf16(a1, b, acc[1][t]);                               \
      }                                                                        \
    }                                                                          \
    __syncthreads();                                                           \
  }

// --- QKV GEMM + head-split scatter ------------------------------------------
__global__ void qkv_gemm_kernel(const __bf16* __restrict__ xb,
                                const __bf16* __restrict__ wT,
                                const float*  __restrict__ bias,
                                __bf16* __restrict__ qb,
                                __bf16* __restrict__ kb,
                                __bf16* __restrict__ vT) {
  GEMM_PROLOGUE(xb, wT)

#pragma unroll
  for (int mi = 0; mi < 2; ++mi) {
#pragma unroll
    for (int t = 0; t < 4; ++t) {
      int n = nBaseBlk + nBase + t * 16 + (lane & 15);
      float bn = bias[n];
      int which = n >> 10;                 // 0=q 1=k 2=v
      int c = n & (EMB - 1);
      int h = c >> 6, d = c & 63;
#pragma unroll
      for (int r = 0; r < 8; ++r) {
        int m  = mBase + mi * 16 + r + half * 8;
        int bb = m >> 11, tt = m & (SEQ - 1);
        __bf16 val = (__bf16)(acc[mi][t][r] + bn);
        if (which == 0)
          qb[((size_t)(bb * NH + h) * SEQ + tt) * HD + d] = val;
        else if (which == 1)
          kb[((size_t)(bb * NH + h) * SEQ + tt) * HD + d] = val;
        else
          vT[((size_t)(bb * NH + h) * HD + d) * SEQ + tt] = val;  // transposed
      }
    }
  }
}

// --- Output projection -> fp32 ----------------------------------------------
__global__ void proj_gemm_kernel(const __bf16* __restrict__ yb,
                                 const __bf16* __restrict__ wT,
                                 const float*  __restrict__ bias,
                                 float* __restrict__ out) {
  GEMM_PROLOGUE(yb, wT)

#pragma unroll
  for (int mi = 0; mi < 2; ++mi) {
#pragma unroll
    for (int t = 0; t < 4; ++t) {
      int n = nBaseBlk + nBase + t * 16 + (lane & 15);
      float bn = bias[n];
#pragma unroll
      for (int r = 0; r < 8; ++r) {
        int m = mBase + mi * 16 + r + half * 8;
        out[(size_t)m * EMB + n] = acc[mi][t][r] + bn;
      }
    }
  }
}

// --- Flash attention: one wave per (b, h, 16-row q block) -------------------
#define PLDS 40   /* padded row stride of P stage tile */
__global__ void flash_attn_kernel(const __bf16* __restrict__ qb,
                                  const __bf16* __restrict__ kb,
                                  const __bf16* __restrict__ vT,
                                  __bf16* __restrict__ yb) {
  __shared__ __attribute__((aligned(16))) __bf16 pstage[8][16 * PLDS];

  int lane = threadIdx.x & 31;
  int wid  = threadIdx.x >> 5;
  int half = lane >> 4;
  int gw   = blockIdx.x * 8 + wid;     // 8192 waves total
  int qblk = gw & 127;
  int bh   = gw >> 7;
  int h    = bh & 15;
  int b    = bh >> 4;
  int qBase = qblk * 16;
  const float SCALE = 0.125f;          // 1/sqrt(64)

  const __bf16* qp = qb + ((size_t)bh * SEQ + qBase) * HD;
  v16bf qf0 = load_a_frag(qp, HD, lane);       // d 0..31
  v16bf qf1 = load_a_frag(qp + 32, HD, lane);  // d 32..63

  v8f o0 = {}, o1 = {}, o2 = {}, o3 = {};
  float mst[8], lst[8];
#pragma unroll
  for (int r = 0; r < 8; ++r) { mst[r] = -1e30f; lst[r] = 0.0f; }

  __bf16* pb = pstage[wid];
  int nkb = (qBase + 16 + 31) >> 5;

  for (int kbI = 0; kbI < nkb; ++kbI) {
    int kBase = kbI * 32;
    // S = Q K^T over the 32-key block (two 16x16 tiles)
    const __bf16* kp = kb + ((size_t)bh * SEQ + kBase) * HD;
    v8f s0 = {}, s1 = {};
    s0 = wmma_bf16(qf0, load_b_frag(kp, HD, lane), s0);
    s0 = wmma_bf16(qf1, load_b_frag(kp + 32, HD, lane), s0);
    s1 = wmma_bf16(qf0, load_b_frag(kp + 16 * HD, HD, lane), s1);
    s1 = wmma_bf16(qf1, load_b_frag(kp + 16 * HD + 32, HD, lane), s1);

    // issue V fragment loads now; the softmax VALU below hides their latency
    const __bf16* vp = vT + (size_t)bh * HD * SEQ + kBase;
    v16bf bv0 = load_b_frag(vp + 0 * 16 * SEQ, SEQ, lane);
    v16bf bv1 = load_b_frag(vp + 1 * 16 * SEQ, SEQ, lane);
    v16bf bv2 = load_b_frag(vp + 2 * 16 * SEQ, SEQ, lane);
    v16bf bv3 = load_b_frag(vp + 3 * 16 * SEQ, SEQ, lane);

    int n0 = kBase + (lane & 15);
    int n1 = n0 + 16;
#pragma unroll
    for (int r = 0; r < 8; ++r) {
      int qrow = qBase + r + half * 8;
      float v0 = (n0 > qrow) ? -1e30f : s0[r] * SCALE;
      float v1 = (n1 > qrow) ? -1e30f : s1[r] * SCALE;
      float mloc = rowred_max(fmaxf(v0, v1));
      float mnew = fmaxf(mst[r], mloc);
      float alpha = __expf(mst[r] - mnew);
      float p0 = __expf(v0 - mnew);
      float p1 = __expf(v1 - mnew);
      lst[r] = lst[r] * alpha + rowred_sum(p0 + p1);
      mst[r] = mnew;
      o0[r] *= alpha; o1[r] *= alpha; o2[r] *= alpha; o3[r] *= alpha;
      int mrow = r + half * 8;
      pb[mrow * PLDS + (lane & 15)]      = (__bf16)p0;
      pb[mrow * PLDS + 16 + (lane & 15)] = (__bf16)p1;
    }
    __builtin_amdgcn_wave_barrier();   // same-wave DS ops are in-order
    v16bf pa = load_a_frag(pb, PLDS, lane);
    __builtin_amdgcn_wave_barrier();

    o0 = wmma_bf16(pa, bv0, o0);
    o1 = wmma_bf16(pa, bv1, o1);
    o2 = wmma_bf16(pa, bv2, o2);
    o3 = wmma_bf16(pa, bv3, o3);
  }

#pragma unroll
  for (int r = 0; r < 8; ++r) {
    float inv = 1.0f / lst[r];
    int qrow = qBase + r + half * 8;
    size_t base = ((size_t)b * SEQ + qrow) * EMB + h * HD + (lane & 15);
    yb[base +  0] = (__bf16)(o0[r] * inv);
    yb[base + 16] = (__bf16)(o1[r] * inv);
    yb[base + 32] = (__bf16)(o2[r] * inv);
    yb[base + 48] = (__bf16)(o3[r] * inv);
  }
}

// ---------------------------------------------------------------------------
extern "C" void kernel_launch(void* const* d_in, const int* in_sizes, int n_in,
                              void* d_out, int out_size, void* d_ws, size_t ws_size,
                              hipStream_t stream) {
  (void)in_sizes; (void)n_in; (void)out_size; (void)ws_size;
  const float* x      = (const float*)d_in[0];
  const float* W_qkv  = (const float*)d_in[1];
  const float* b_qkv  = (const float*)d_in[2];
  const float* W_proj = (const float*)d_in[3];
  const float* b_proj = (const float*)d_in[4];
  float* out = (float*)d_out;

  char* ws = (char*)d_ws;
  __bf16* xb     = (__bf16*)ws;  ws += (size_t)MTOT * EMB * 2;
  __bf16* wqkvT  = (__bf16*)ws;  ws += (size_t)NQKV * EMB * 2;
  __bf16* wprojT = (__bf16*)ws;  ws += (size_t)EMB * EMB * 2;
  __bf16* qb     = (__bf16*)ws;  ws += (size_t)MTOT * EMB * 2;
  __bf16* kb     = (__bf16*)ws;  ws += (size_t)MTOT * EMB * 2;
  __bf16* vT     = (__bf16*)ws;  ws += (size_t)MTOT * EMB * 2;
  __bf16* yb     = (__bf16*)ws;

  cvt_x_kernel<<<(MTOT * EMB) / 256, 256, 0, stream>>>(x, xb, MTOT * EMB);
  cvt_wqkv_kernel<<<(NQKV * EMB) / 256, 256, 0, stream>>>(W_qkv, wqkvT);
  cvt_wproj_kernel<<<(EMB * EMB) / 256, 256, 0, stream>>>(W_proj, wprojT);

  qkv_gemm_kernel<<<dim3(NQKV / 128, MTOT / 128), 256, 0, stream>>>(
      xb, wqkvT, b_qkv, qb, kb, vT);

  // B*H*(T/16) = 8192 waves / 8 per block = 1024 blocks
  flash_attn_kernel<<<1024, 256, 0, stream>>>(qb, kb, vT, yb);

  proj_gemm_kernel<<<dim3(EMB / 128, MTOT / 128), 256, 0, stream>>>(
      yb, wprojT, b_proj, out);
}
